// GATE_79207786873630
// MI455X (gfx1250) — matrix operand order repaired and checked by reading, not compiled, not run on hardware
//
#include <hip/hip_runtime.h>
#include <hip/hip_bf16.h>

typedef _Float16 f16;
typedef __attribute__((ext_vector_type(16))) _Float16 v16h;
typedef __attribute__((ext_vector_type(8)))  _Float16 h8;
typedef __attribute__((ext_vector_type(8)))  float    v8f;

union Frag16 { v16h v; h8 h[2]; };

// ---------------- helpers ----------------

// monotonic float<->uint encoding for atomic segment-max
__device__ __forceinline__ unsigned enc_f32(float f) {
  unsigned u = __float_as_uint(f);
  return (u & 0x80000000u) ? ~u : (u | 0x80000000u);
}
__device__ __forceinline__ float dec_f32(unsigned u) {
  return (u & 0x80000000u) ? __uint_as_float(u ^ 0x80000000u)
                           : __uint_as_float(~u);
}

// ---------------- precision conversion ----------------

__global__ void gat_cvt_f16(const float* __restrict__ a, f16* __restrict__ b,
                            size_t n) {
  size_t i = (size_t)blockIdx.x * blockDim.x + threadIdx.x;
  if (i < n) b[i] = (f16)a[i];
}

// f32 [K x N] row-major  ->  f16 [N x K] (transposed, so GEMM B-fragment
// loads become two contiguous 16B vector loads per lane)
__global__ void gat_cvt_f16_t(const float* __restrict__ in,
                              f16* __restrict__ out, int K, int N) {
  int i = blockIdx.x * blockDim.x + threadIdx.x;
  if (i >= K * N) return;
  int k = i / N, n = i % N;
  out[(size_t)n * K + k] = (f16)in[i];
}

// ---------------- WMMA GEMM: C[MxN] = A[MxK] @ B[KxN] ----------------
// A f16 row-major [M x K]; Bt f16 = B transposed [N x K]; C f32 row-major.
// M%16==0; K,N compile-time (K%32==0, N%16==0).
// block = 128 threads (4 waves); wave -> one 16x16 C tile; K-loop fully
// unrolled -> back-to-back v_wmma with b128 operand loads.

template <int K, int N>
__global__ __launch_bounds__(128) void gat_gemm_wmma(
    const f16* __restrict__ A, const f16* __restrict__ Bt,
    float* __restrict__ C) {
  const int wid  = threadIdx.x >> 5;
  const int lane = threadIdx.x & 31;
  const int mTile = blockIdx.x;
  const int nTile = blockIdx.y * 4 + wid;
  if (nTile * 16 >= N) return;

  const int half = lane >> 4;   // 0: lanes 0-15, 1: lanes 16-31
  const int l16  = lane & 15;

  // A fragment (16-bit 16x32 layout): elems 0-7 -> k=k0+half*8+e,
  //                                   elems 8-15 -> k=k0+16+half*8+(e-8)
  const f16* arow = A  + (size_t)(mTile * 16 + l16) * K + half * 8;
  // B fragment (32x16): lane = column n, same k order as A fragment
  const f16* brow = Bt + (size_t)(nTile * 16 + l16) * K + half * 8;

  v8f acc = {};
#pragma unroll
  for (int k0 = 0; k0 < K; k0 += 32) {
    Frag16 a, b;
    a.h[0] = *(const h8*)(arow + k0);
    a.h[1] = *(const h8*)(arow + k0 + 16);
    b.h[0] = *(const h8*)(brow + k0);
    b.h[1] = *(const h8*)(brow + k0 + 16);
    acc = __builtin_amdgcn_wmma_f32_16x16x32_f16(
        /*neg_a=*/false, a.v, /*neg_b=*/false, b.v,
        /*c_mod=*/(short)0, acc, /*reuse_a=*/false, /*reuse_b=*/false);
  }

  // C layout: VGPR r -> row (r + 8*half), lane -> column n
  float* crow = C + (size_t)(mTile * 16 + half * 8) * N + nTile * 16 + l16;
#pragma unroll
  for (int r = 0; r < 8; ++r) crow[(size_t)r * N] = acc[r];
}

// ---------------- layer-1 edge kernels (H=4, C=64, F=256) ----------------
// one wave per edge; lane covers 8 of 256 features (i = lane + 32*j, h = j>>1)

__global__ __launch_bounds__(256) void gat_edge_score_l1(
    const float* __restrict__ xl, const float* __restrict__ xr,
    const float* __restrict__ att, const int* __restrict__ ei,
    float* __restrict__ score, unsigned* __restrict__ smax, int E, int Etot) {
  int e = blockIdx.x * 8 + (threadIdx.x >> 5);
  if (e >= Etot) return;
  int lane = threadIdx.x & 31;
  int s, d;
  if (e < E) { s = ei[e]; d = ei[E + e]; } else { s = d = e - E; }
  const float* ps = xl + (size_t)s * 256;
  const float* pd = xr + (size_t)d * 256;
  float p[4] = {0.f, 0.f, 0.f, 0.f};
#pragma unroll
  for (int j = 0; j < 8; ++j) {
    int i = lane + 32 * j;
    float v = ps[i] + pd[i];
    v = v > 0.f ? v : 0.2f * v;              // LeakyReLU(0.2)
    p[j >> 1] += v * att[i];                 // att flat index = h*64 + c
  }
#pragma unroll
  for (int h = 0; h < 4; ++h)
#pragma unroll
    for (int off = 16; off; off >>= 1)
      p[h] += __shfl_xor(p[h], off, 32);
  if (lane == 0) {
#pragma unroll
    for (int h = 0; h < 4; ++h) {
      score[(size_t)e * 4 + h] = p[h];
      atomicMax(&smax[(size_t)d * 4 + h], enc_f32(p[h]));
    }
  }
}

__global__ void gat_edge_exp_l1(float* __restrict__ score,
                                const unsigned* __restrict__ smax,
                                float* __restrict__ denom,
                                const int* __restrict__ ei, int E, int Etot) {
  int e = blockIdx.x * blockDim.x + threadIdx.x;
  if (e >= Etot) return;
  int d = (e < E) ? ei[E + e] : e - E;
#pragma unroll
  for (int h = 0; h < 4; ++h) {
    float m = dec_f32(smax[(size_t)d * 4 + h]);
    float ex = __expf(score[(size_t)e * 4 + h] - m);
    score[(size_t)e * 4 + h] = ex;
    atomicAdd(&denom[(size_t)d * 4 + h], ex);
  }
}

__global__ __launch_bounds__(256) void gat_edge_agg_l1(
    const float* __restrict__ xl, const float* __restrict__ score,
    const float* __restrict__ denom, const int* __restrict__ ei,
    float* __restrict__ out, int E, int Etot) {
  int e = blockIdx.x * 8 + (threadIdx.x >> 5);
  if (e >= Etot) return;
  int lane = threadIdx.x & 31;
  int s, d;
  if (e < E) { s = ei[e]; d = ei[E + e]; } else { s = d = e - E; }
  const float4 ex = *(const float4*)(score + (size_t)e * 4);
  const float4 dn = *(const float4*)(denom + (size_t)d * 4);
  float al[4] = {ex.x / dn.x, ex.y / dn.y, ex.z / dn.z, ex.w / dn.w};
  const float* ps = xl + (size_t)s * 256;
  float* po = out + (size_t)d * 256;
#pragma unroll
  for (int j = 0; j < 8; ++j) {
    int i = lane + 32 * j;
    atomicAdd(&po[i], ps[i] * al[j >> 1]);
  }
}

// h = elu(out + bias); store as f16 for the layer-2 WMMA GEMM
__global__ void gat_bias_elu_cvt(const float* __restrict__ in,
                                 const float* __restrict__ bias,
                                 f16* __restrict__ outh, size_t n, int C) {
  size_t i = (size_t)blockIdx.x * blockDim.x + threadIdx.x;
  if (i >= n) return;
  float v = in[i] + bias[i % C];
  v = v > 0.f ? v : (__expf(v) - 1.f);       // ELU
  outh[i] = (f16)v;
}

// ---------------- layer-2 edge kernels (H=1, C=32) ----------------

__global__ __launch_bounds__(256) void gat_edge_score_l2(
    const float* __restrict__ xl, const float* __restrict__ xr,
    const float* __restrict__ att, const int* __restrict__ ei,
    float* __restrict__ score, unsigned* __restrict__ smax, int E, int Etot) {
  int e = blockIdx.x * 8 + (threadIdx.x >> 5);
  if (e >= Etot) return;
  int lane = threadIdx.x & 31;
  int s, d;
  if (e < E) { s = ei[e]; d = ei[E + e]; } else { s = d = e - E; }
  float v = xl[(size_t)s * 32 + lane] + xr[(size_t)d * 32 + lane];
  v = v > 0.f ? v : 0.2f * v;
  float p = v * att[lane];
#pragma unroll
  for (int off = 16; off; off >>= 1) p += __shfl_xor(p, off, 32);
  if (lane == 0) {
    score[e] = p;
    atomicMax(&smax[d], enc_f32(p));
  }
}

__global__ void gat_edge_exp_l2(float* __restrict__ score,
                                const unsigned* __restrict__ smax,
                                float* __restrict__ denom,
                                const int* __restrict__ ei, int E, int Etot) {
  int e = blockIdx.x * blockDim.x + threadIdx.x;
  if (e >= Etot) return;
  int d = (e < E) ? ei[E + e] : e - E;
  float ex = __expf(score[e] - dec_f32(smax[d]));
  score[e] = ex;
  atomicAdd(&denom[d], ex);
}

__global__ __launch_bounds__(256) void gat_edge_agg_l2(
    const float* __restrict__ xl, const float* __restrict__ score,
    const float* __restrict__ denom, const int* __restrict__ ei,
    float* __restrict__ out, int E, int Etot) {
  int e = blockIdx.x * 8 + (threadIdx.x >> 5);
  if (e >= Etot) return;
  int lane = threadIdx.x & 31;
  int s, d;
  if (e < E) { s = ei[e]; d = ei[E + e]; } else { s = d = e - E; }
  float alpha = score[e] / denom[d];
  atomicAdd(&out[(size_t)d * 32 + lane],
            xl[(size_t)s * 32 + lane] * alpha);
}

__global__ void gat_z_bias(float* __restrict__ z, const float* __restrict__ bias,
                           size_t n, int C) {
  size_t i = (size_t)blockIdx.x * blockDim.x + threadIdx.x;
  if (i < n) z[i] += bias[i % C];
}

// pred[e] = sigmoid(<z[row], z[col]>) over original edges
__global__ __launch_bounds__(256) void gat_decode(
    const float* __restrict__ z, const int* __restrict__ ei,
    float* __restrict__ pred, int E) {
  int e = blockIdx.x * 8 + (threadIdx.x >> 5);
  if (e >= E) return;
  int lane = threadIdx.x & 31;
  int r = ei[e], c = ei[E + e];
  float p = z[(size_t)r * 32 + lane] * z[(size_t)c * 32 + lane];
#pragma unroll
  for (int off = 16; off; off >>= 1) p += __shfl_xor(p, off, 32);
  if (lane == 0) pred[e] = 1.f / (1.f + __expf(-p));
}

// ---------------- launch ----------------

extern "C" void kernel_launch(void* const* d_in, const int* in_sizes, int n_in,
                              void* d_out, int out_size, void* d_ws, size_t ws_size,
                              hipStream_t stream) {
  const float* x     = (const float*)d_in[0];
  const int*   ei    = (const int*)d_in[1];
  const float* W_l1  = (const float*)d_in[2];
  const float* W_r1  = (const float*)d_in[3];
  const float* att1  = (const float*)d_in[4];
  const float* bias1 = (const float*)d_in[5];
  const float* W_l2  = (const float*)d_in[6];
  const float* W_r2  = (const float*)d_in[7];
  const float* att2  = (const float*)d_in[8];
  const float* bias2 = (const float*)d_in[9];
  float* pred = (float*)d_out;

  const int IN = 128, F1 = 256, LAT = 32;       // F1 = HEADS*HID
  const int Nn = in_sizes[0] / IN;              // 50000
  const int E  = in_sizes[1] / 2;               // 800000
  const int Et = E + Nn;                        // with self-loops

  // workspace carve-up (256B aligned)
  char* base = (char*)d_ws;
  size_t off = 0;
  auto alloc = [&](size_t bytes) {
    void* p = base + off;
    off = (off + bytes + 255) & ~(size_t)255;
    return p;
  };
  f16*      xf16   = (f16*)     alloc((size_t)Nn * IN * sizeof(f16));
  f16*      wl1t   = (f16*)     alloc((size_t)IN * F1 * sizeof(f16));  // [F1 x IN]
  f16*      wr1t   = (f16*)     alloc((size_t)IN * F1 * sizeof(f16));
  float*    xl1    = (float*)   alloc((size_t)Nn * F1 * sizeof(float));
  float*    xr1    = (float*)   alloc((size_t)Nn * F1 * sizeof(float));
  float*    score1 = (float*)   alloc((size_t)Et * 4 * sizeof(float));
  unsigned* smax1  = (unsigned*)alloc((size_t)Nn * 4 * sizeof(unsigned));
  float*    denom1 = (float*)   alloc((size_t)Nn * 4 * sizeof(float));
  float*    out1   = (float*)   alloc((size_t)Nn * F1 * sizeof(float));
  f16*      hf16   = (f16*)     alloc((size_t)Nn * F1 * sizeof(f16));
  f16*      wl2t   = (f16*)     alloc((size_t)F1 * LAT * sizeof(f16)); // [LAT x F1]
  f16*      wr2t   = (f16*)     alloc((size_t)F1 * LAT * sizeof(f16));
  float*    xl2    = (float*)   alloc((size_t)Nn * LAT * sizeof(float));
  float*    xr2    = (float*)   alloc((size_t)Nn * LAT * sizeof(float));
  float*    score2 = (float*)   alloc((size_t)Et * sizeof(float));
  unsigned* smax2  = (unsigned*)alloc((size_t)Nn * sizeof(unsigned));
  float*    denom2 = (float*)   alloc((size_t)Nn * sizeof(float));
  float*    out2   = (float*)   alloc((size_t)Nn * LAT * sizeof(float));
  (void)ws_size; (void)n_in;

  // zero accumulators (graph-capture-legal, re-done every call)
  hipMemsetAsync(smax1,  0, (size_t)Nn * 4 * sizeof(unsigned), stream);
  hipMemsetAsync(denom1, 0, (size_t)Nn * 4 * sizeof(float), stream);
  hipMemsetAsync(out1,   0, (size_t)Nn * F1 * sizeof(float), stream);
  hipMemsetAsync(smax2,  0, (size_t)Nn * sizeof(unsigned), stream);
  hipMemsetAsync(denom2, 0, (size_t)Nn * sizeof(float), stream);
  hipMemsetAsync(out2,   0, (size_t)Nn * LAT * sizeof(float), stream);

  auto blks = [](size_t n, int b) { return (unsigned)((n + b - 1) / b); };

  // f32 -> f16 conversions (weights transposed for vectorized B fragments)
  gat_cvt_f16<<<blks((size_t)Nn * IN, 256), 256, 0, stream>>>(x, xf16, (size_t)Nn * IN);
  gat_cvt_f16_t<<<blks((size_t)IN * F1, 256), 256, 0, stream>>>(W_l1, wl1t, IN, F1);
  gat_cvt_f16_t<<<blks((size_t)IN * F1, 256), 256, 0, stream>>>(W_r1, wr1t, IN, F1);

  // layer-1 GEMMs: [Nn x 128] @ [128 x 256]  (Nn % 16 == 0)
  {
    dim3 grid(Nn / 16, (256 / 16 + 3) / 4);
    gat_gemm_wmma<128, 256><<<grid, 128, 0, stream>>>(xf16, wl1t, xl1);
    gat_gemm_wmma<128, 256><<<grid, 128, 0, stream>>>(xf16, wr1t, xr1);
  }

  // layer-1 attention (segment softmax over dst) + aggregation
  gat_edge_score_l1<<<blks(Et, 8), 256, 0, stream>>>(xl1, xr1, att1, ei, score1, smax1, E, Et);
  gat_edge_exp_l1<<<blks(Et, 256), 256, 0, stream>>>(score1, smax1, denom1, ei, E, Et);
  gat_edge_agg_l1<<<blks(Et, 8), 256, 0, stream>>>(xl1, score1, denom1, ei, out1, E, Et);

  // h = elu(out1 + bias1), stored f16 for layer-2 WMMA
  gat_bias_elu_cvt<<<blks((size_t)Nn * F1, 256), 256, 0, stream>>>(
      out1, bias1, hf16, (size_t)Nn * F1, F1);

  gat_cvt_f16_t<<<blks((size_t)F1 * LAT, 256), 256, 0, stream>>>(W_l2, wl2t, F1, LAT);
  gat_cvt_f16_t<<<blks((size_t)F1 * LAT, 256), 256, 0, stream>>>(W_r2, wr2t, F1, LAT);

  // layer-2 GEMMs: [Nn x 256] @ [256 x 32]
  {
    dim3 grid(Nn / 16, 1);  // 2 n-tiles; waves 2,3 exit early
    gat_gemm_wmma<256, 32><<<grid, 128, 0, stream>>>(hf16, wl2t, xl2);
    gat_gemm_wmma<256, 32><<<grid, 128, 0, stream>>>(hf16, wr2t, xr2);
  }

  // layer-2 attention + aggregation (H=1)
  gat_edge_score_l2<<<blks(Et, 8), 256, 0, stream>>>(xl2, xr2, att2, ei, score2, smax2, E, Et);
  gat_edge_exp_l2<<<blks(Et, 256), 256, 0, stream>>>(score2, smax2, denom2, ei, E, Et);
  gat_edge_agg_l2<<<blks(Et, 8), 256, 0, stream>>>(xl2, score2, denom2, ei, out2, E, Et);

  // z = out2 + bias2 (heads=1, concat=False -> mean is identity)
  gat_z_bias<<<blks((size_t)Nn * LAT, 256), 256, 0, stream>>>(out2, bias2, (size_t)Nn * LAT, LAT);

  // pred = sigmoid(<z[row], z[col]>) over original edges
  gat_decode<<<blks(E, 8), 256, 0, stream>>>(out2, ei, pred, E);
  (void)out_size;
}
